// CharRNN_39324720562176
// MI455X (gfx1250) — compile-verified, matrix operand order checked by reading, not run on hardware
//
#include <hip/hip_runtime.h>
#include <hip/hip_bf16.h>

// ---------------- problem dims ----------------
#define V_  32000
#define B_  32
#define T_  128
#define R_  1024
#define L_  4
#define R2  (2*R_)      // 2048
#define R4  (4*R_)      // 4096

typedef __attribute__((ext_vector_type(16))) __bf16 v16bf;
typedef __attribute__((ext_vector_type(8)))  float  v8f;

__device__ __forceinline__ unsigned short f2bf(float f) {
    unsigned int u = __builtin_bit_cast(unsigned int, f);
    u += 0x7FFFu + ((u >> 16) & 1u);          // round-to-nearest-even
    return (unsigned short)(u >> 16);
}
__device__ __forceinline__ float sigmoidf_(float x) {
    return 1.0f / (1.0f + __expf(-x));
}
__device__ __forceinline__ void load_frag(v16bf& f, const unsigned short* p0,
                                          const unsigned short* p1) {
    ((uint4*)&f)[0] = *(const uint4*)p0;
    ((uint4*)&f)[1] = *(const uint4*)p1;
}

// ---------------- prep kernels ----------------

// W[l] : [2R, 4R] f32  ->  Wt[l] : [4R, 2R] bf16 (N-major, K contiguous)
__global__ void transpose_w_kernel(const float* __restrict__ W,
                                   unsigned short* __restrict__ Wt) {
    __shared__ float tile[32][33];
    const int l  = blockIdx.z;
    const int nb = blockIdx.x * 32;   // column block of W (N)
    const int kb = blockIdx.y * 32;   // row block of W (K)
    const float* Wl = W + (size_t)l * R2 * R4;
    unsigned short* Wtl = Wt + (size_t)l * R4 * R2;
    const int tx = threadIdx.x, ty = threadIdx.y;   // (32, 8)
    #pragma unroll
    for (int i = 0; i < 32; i += 8)
        tile[ty + i][tx] = Wl[(size_t)(kb + ty + i) * R4 + nb + tx];
    __syncthreads();
    #pragma unroll
    for (int i = 0; i < 32; i += 8)
        Wtl[(size_t)(nb + ty + i) * R2 + kb + tx] = f2bf(tile[tx][ty + i]);
}

// elementwise f32 -> bf16 (softmax_w is already N-major for WMMA B)
__global__ void convert_bf16_kernel(const float* __restrict__ src,
                                    unsigned short* __restrict__ dst,
                                    long long n) {
    long long i = (long long)blockIdx.x * blockDim.x + threadIdx.x;
    const long long stride = (long long)gridDim.x * blockDim.x;
    for (; i < n; i += stride) dst[i] = f2bf(src[i]);
}

// xs[t*B*R + b*R + r] = bf16(embedding[input[b*T + t]][r])
__global__ void embed_kernel(const int* __restrict__ input_data,
                             const float* __restrict__ embedding,
                             unsigned short* __restrict__ xs) {
    const long long idx = (long long)blockIdx.x * blockDim.x + threadIdx.x;
    const int r = (int)(idx & (R_ - 1));
    const int b = (int)((idx >> 10) & (B_ - 1));
    const int t = (int)(idx >> 15);
    const int tok = input_data[b * T_ + t];
    xs[idx] = f2bf(embedding[(size_t)tok * R_ + r]);
}

__global__ void init_state_kernel(float* __restrict__ c,
                                  unsigned short* __restrict__ h) {
    const int idx = blockIdx.x * blockDim.x + threadIdx.x;   // B*R = 32768
    c[idx] = 0.0f;
    h[idx] = 0;
}

// ---------------- LSTM step GEMM: z[B,4R] = [x|h] @ W + bias ----------------
// 256 waves; each wave computes a 32x16 tile (both M tiles) so each B (weight)
// fragment is loaded ONCE and used by two WMMAs -> halves weight traffic.
// K loop split: first 1024 from x_t, second 1024 from h (no per-iter selects).
__global__ void lstm_gemm_kernel(const unsigned short* __restrict__ x_t,  // [B,R] bf16
                                 const unsigned short* __restrict__ h,    // [B,R] bf16
                                 const unsigned short* __restrict__ Wt,   // [4R,2R] bf16
                                 const float* __restrict__ bias,          // [4R]
                                 float* __restrict__ z) {                 // [B,4R]
    const int nt   = (blockIdx.x * blockDim.x + threadIdx.x) >> 5;  // 0..255
    const int lane = threadIdx.x & 31;
    const int half = lane >> 4;
    const int row  = lane & 15;

    const unsigned short* a0x = x_t + (size_t)row * R_;         // M tile 0 rows
    const unsigned short* a1x = x_t + (size_t)(16 + row) * R_;  // M tile 1 rows
    const unsigned short* a0h = h   + (size_t)row * R_;
    const unsigned short* a1h = h   + (size_t)(16 + row) * R_;
    const unsigned short* brow = Wt + (size_t)(nt * 16 + row) * R2;

    v8f acc0 = {}, acc1 = {};
    // ---- K = 0..1023 : A from x_t, B from Wt[:, 0:1024] ----
    #pragma unroll 4
    for (int ks = 0; ks < 32; ++ks) {
        const int k0 = ks * 32 + half * 8;
        const int k1 = k0 + 16;
        v16bf a0, a1, b;
        load_frag(b,  brow + k0, brow + k1);
        load_frag(a0, a0x + k0,  a0x + k1);
        load_frag(a1, a1x + k0,  a1x + k1);
        __builtin_prefetch(brow + k0 + 512, 0, 1);   // global_prefetch_b8
        acc0 = __builtin_amdgcn_wmma_f32_16x16x32_bf16(
                   false, a0, false, b, (short)0, acc0, false, false);
        acc1 = __builtin_amdgcn_wmma_f32_16x16x32_bf16(
                   false, a1, false, b, (short)0, acc1, false, false);
    }
    // ---- K = 1024..2047 : A from h, B from Wt[:, 1024:2048] ----
    const unsigned short* brow2 = brow + R_;
    #pragma unroll 4
    for (int ks = 0; ks < 32; ++ks) {
        const int k0 = ks * 32 + half * 8;
        const int k1 = k0 + 16;
        v16bf a0, a1, b;
        load_frag(b,  brow2 + k0, brow2 + k1);
        load_frag(a0, a0h + k0,   a0h + k1);
        load_frag(a1, a1h + k0,   a1h + k1);
        __builtin_prefetch(brow2 + k0 + 512, 0, 1);
        acc0 = __builtin_amdgcn_wmma_f32_16x16x32_bf16(
                   false, a0, false, b, (short)0, acc0, false, false);
        acc1 = __builtin_amdgcn_wmma_f32_16x16x32_bf16(
                   false, a1, false, b, (short)0, acc1, false, false);
    }
    // D layout: VGPR v -> M = v + 8*half, N = row
    const int nOut = nt * 16 + row;
    const float bv = bias[nOut];
    #pragma unroll
    for (int v = 0; v < 8; ++v) {
        const int m0 = half * 8 + v;
        z[(size_t)m0 * R4 + nOut]        = acc0[v] + bv;
        z[(size_t)(16 + m0) * R4 + nOut] = acc1[v] + bv;
    }
}

// ---------------- LSTM cell: 5x LayerNorm + gates; one block per batch row ---
__device__ __forceinline__ void blk_reduce2(float& s, float& q, float* sm) {
    #pragma unroll
    for (int off = 16; off; off >>= 1) {
        s += __shfl_down(s, off, 32);
        q += __shfl_down(q, off, 32);
    }
    const int wid = threadIdx.x >> 5;
    if ((threadIdx.x & 31) == 0) { sm[wid] = s; sm[8 + wid] = q; }
    __syncthreads();
    s = 0.0f; q = 0.0f;
    #pragma unroll
    for (int w = 0; w < 8; ++w) { s += sm[w]; q += sm[8 + w]; }
    __syncthreads();
}

__global__ void lstm_cell_kernel(const float* __restrict__ z,     // [B,4R]
                                 const float* __restrict__ ln_g,  // [5,R]
                                 const float* __restrict__ ln_b,  // [5,R]
                                 float* __restrict__ c,           // [B,R]
                                 unsigned short* __restrict__ h,  // [B,R] bf16
                                 unsigned short* __restrict__ y) {// [B,R] bf16 (layer out @ t)
    __shared__ float sm[16];
    const int b   = blockIdx.x;
    const int tid = threadIdx.x;      // 256 threads, 4 elems each
    const float* zb = z + (size_t)b * R4;

    float g[4][4];
    #pragma unroll
    for (int gate = 0; gate < 4; ++gate) {
        float v[4], s = 0.0f, q = 0.0f;
        #pragma unroll
        for (int j = 0; j < 4; ++j) {
            const int r = tid + j * 256;
            const float x = zb[gate * R_ + r];
            v[j] = x; s += x; q += x * x;
        }
        blk_reduce2(s, q, sm);
        const float mean = s * (1.0f / R_);
        const float var  = q * (1.0f / R_) - mean * mean;
        const float inv  = rsqrtf(var + 1e-5f);
        #pragma unroll
        for (int j = 0; j < 4; ++j) {
            const int r = tid + j * 256;
            g[gate][j] = (v[j] - mean) * inv * ln_g[gate * R_ + r] + ln_b[gate * R_ + r];
        }
    }
    // c_new = c * sigmoid(f + 1) + sigmoid(i) * tanh(j)
    float cv[4], s = 0.0f, q = 0.0f;
    #pragma unroll
    for (int j = 0; j < 4; ++j) {
        const int r = tid + j * 256;
        const float ii = sigmoidf_(g[0][j]);
        const float jj = tanhf(g[1][j]);
        const float ff = sigmoidf_(g[2][j] + 1.0f);
        const float cn = c[(size_t)b * R_ + r] * ff + ii * jj;
        cv[j] = cn;
        c[(size_t)b * R_ + r] = cn;
        s += cn; q += cn * cn;
    }
    blk_reduce2(s, q, sm);
    const float mean = s * (1.0f / R_);
    const float var  = q * (1.0f / R_) - mean * mean;
    const float inv  = rsqrtf(var + 1e-5f);
    #pragma unroll
    for (int j = 0; j < 4; ++j) {
        const int r = tid + j * 256;
        const float lc = (cv[j] - mean) * inv * ln_g[4 * R_ + r] + ln_b[4 * R_ + r];
        const float hn = tanhf(lc) * sigmoidf_(g[3][j]);
        const unsigned short hb = f2bf(hn);
        h[(size_t)b * R_ + r] = hb;
        y[(size_t)b * R_ + r] = hb;
    }
}

// ---------------- logits GEMM: [B*T, R] @ softmax_w^T + softmax_b ----------
// 128 M-groups (32 rows) x 1000 N-pairs (32 cols); each wave computes a 32x32
// register tile: 2 A frags x 2 B frags -> 4 WMMAs per 4 fragment loads.
__global__ void logits_gemm_kernel(const unsigned short* __restrict__ out,  // [T,B,R] bf16
                                   const unsigned short* __restrict__ swb,  // [V,R] bf16
                                   const float* __restrict__ softmax_b,     // [V]
                                   float* __restrict__ logits) {            // [B*T,V]
    const int wave = (blockIdx.x * blockDim.x + threadIdx.x) >> 5;
    const int lane = threadIdx.x & 31;
    const int npair = wave % 1000;
    const int mg    = wave / 1000;       // 0..127
    const int half = lane >> 4;
    const int row  = lane & 15;

    // A rows: logical row m = b*T + t ; out is [T, B, R]
    const int m0 = mg * 32 + row;
    const int m1 = m0 + 16;
    const unsigned short* arow0 =
        out + (size_t)(m0 & (T_ - 1)) * B_ * R_ + (size_t)(m0 >> 7) * R_;
    const unsigned short* arow1 =
        out + (size_t)(m1 & (T_ - 1)) * B_ * R_ + (size_t)(m1 >> 7) * R_;
    const int n0 = npair * 32;
    const unsigned short* b0row = swb + (size_t)(n0 + row) * R_;
    const unsigned short* b1row = swb + (size_t)(n0 + 16 + row) * R_;

    v8f acc00 = {}, acc01 = {}, acc10 = {}, acc11 = {};
    #pragma unroll 2
    for (int ks = 0; ks < 32; ++ks) {
        const int k0 = ks * 32 + half * 8;
        const int k1 = k0 + 16;
        v16bf a0, a1, b0, b1;
        load_frag(a0, arow0 + k0, arow0 + k1);
        load_frag(a1, arow1 + k0, arow1 + k1);
        load_frag(b0, b0row + k0, b0row + k1);
        load_frag(b1, b1row + k0, b1row + k1);
        __builtin_prefetch(b0row + k0 + 256, 0, 1);
        __builtin_prefetch(b1row + k0 + 256, 0, 1);
        acc00 = __builtin_amdgcn_wmma_f32_16x16x32_bf16(
                    false, a0, false, b0, (short)0, acc00, false, false);
        acc01 = __builtin_amdgcn_wmma_f32_16x16x32_bf16(
                    false, a0, false, b1, (short)0, acc01, false, false);
        acc10 = __builtin_amdgcn_wmma_f32_16x16x32_bf16(
                    false, a1, false, b0, (short)0, acc10, false, false);
        acc11 = __builtin_amdgcn_wmma_f32_16x16x32_bf16(
                    false, a1, false, b1, (short)0, acc11, false, false);
    }
    const float sb0 = softmax_b[n0 + row];
    const float sb1 = softmax_b[n0 + 16 + row];
    #pragma unroll
    for (int v = 0; v < 8; ++v) {
        const int mo0 = mg * 32 + half * 8 + v;
        const int mo1 = mo0 + 16;
        logits[(size_t)mo0 * V_ + n0 + row]      = acc00[v] + sb0;
        logits[(size_t)mo0 * V_ + n0 + 16 + row] = acc01[v] + sb1;
        logits[(size_t)mo1 * V_ + n0 + row]      = acc10[v] + sb0;
        logits[(size_t)mo1 * V_ + n0 + 16 + row] = acc11[v] + sb1;
    }
}

// ---------------- host side ----------------
extern "C" void kernel_launch(void* const* d_in, const int* in_sizes, int n_in,
                              void* d_out, int out_size, void* d_ws, size_t ws_size,
                              hipStream_t stream) {
    const int*   input_data = (const int*)  d_in[0];   // [B, T]
    const float* embedding  = (const float*)d_in[1];   // [V, R]
    const float* W          = (const float*)d_in[2];   // [L, 2R, 4R]
    const float* bias       = (const float*)d_in[3];   // [L, 4R]
    const float* ln_g       = (const float*)d_in[4];   // [L, 5, R]
    const float* ln_b       = (const float*)d_in[5];   // [L, 5, R]
    const float* softmax_w  = (const float*)d_in[6];   // [V, R]
    const float* softmax_b  = (const float*)d_in[7];   // [V]
    float* logits = (float*)d_out;                     // [B*T, V]

    // workspace layout (256B aligned)
    char* ws = (char*)d_ws;
    size_t off = 0;
    auto take = [&](size_t bytes) -> char* {
        char* p = ws + off;
        off = (off + bytes + 255) & ~(size_t)255;
        return p;
    };
    unsigned short* Wt  = (unsigned short*)take((size_t)L_ * R4 * R2 * 2); // 64 MB
    unsigned short* swb = (unsigned short*)take((size_t)V_ * R_ * 2);      // 62.5 MB
    unsigned short* xs0 = (unsigned short*)take((size_t)T_ * B_ * R_ * 2); // 8 MB
    unsigned short* xs1 = (unsigned short*)take((size_t)T_ * B_ * R_ * 2); // 8 MB
    float*          z   = (float*)take((size_t)B_ * R4 * 4);
    float*          c   = (float*)take((size_t)B_ * R_ * 4);
    unsigned short* h   = (unsigned short*)take((size_t)B_ * R_ * 2);

    // prep
    transpose_w_kernel<<<dim3(R4 / 32, R2 / 32, L_), dim3(32, 8), 0, stream>>>(W, Wt);
    convert_bf16_kernel<<<4096, 256, 0, stream>>>(softmax_w, swb, (long long)V_ * R_);
    embed_kernel<<<(T_ * B_ * R_) / 256, 256, 0, stream>>>(input_data, embedding, xs0);

    unsigned short* cur = xs0;
    unsigned short* nxt = xs1;
    for (int l = 0; l < L_; ++l) {
        init_state_kernel<<<(B_ * R_) / 256, 256, 0, stream>>>(c, h);
        const unsigned short* Wtl = Wt + (size_t)l * R4 * R2;
        const float* biasl = bias + (size_t)l * R4;
        const float* gl    = ln_g + (size_t)l * 5 * R_;
        const float* bl    = ln_b + (size_t)l * 5 * R_;
        for (int t = 0; t < T_; ++t) {
            // 256 waves = 32 blocks x 8 waves
            lstm_gemm_kernel<<<32, 256, 0, stream>>>(
                cur + (size_t)t * B_ * R_, h, Wtl, biasl, z);
            lstm_cell_kernel<<<B_, 256, 0, stream>>>(
                z, gl, bl, c, h, nxt + (size_t)t * B_ * R_);
        }
        unsigned short* tmp = cur; cur = nxt; nxt = tmp;
    }
    // 128 M-groups * 1000 N-pairs = 128000 waves / 8 per block = 16000 blocks
    logits_gemm_kernel<<<16000, 256, 0, stream>>>(cur, swb, softmax_b, logits);
}
